// RGCN_1709396984332
// MI455X (gfx1250) — matrix operand order repaired and checked by reading, not compiled, not run on hardware
//
#include <hip/hip_runtime.h>

// ---------------- types ----------------
typedef __bf16 bf16_t;
typedef bf16_t v16bf __attribute__((ext_vector_type(16)));
typedef float  v8f   __attribute__((ext_vector_type(8)));

struct U128 { unsigned int a, b, c, d; };   // POD 16-byte chunk

#define R_REL 8
#define D_IN  768
#define D_OUT 768
#define K_SUM (R_REL * D_IN)          // 6144
#define K_AUG (K_SUM + D_IN)          // 6912

#define BM 128
#define BN 128
#define BK 32
#define LDT 40                        // LDS row stride (elems): 80B = 5*16B, bank-spread
#define KT_TOT (K_AUG / BK)           // 216

static __device__ __forceinline__ unsigned int f32_to_bf16(float f) {
    unsigned int u = __builtin_bit_cast(unsigned int, f);
    unsigned int r = u + 0x7FFFu + ((u >> 16) & 1u);   // round-to-nearest-even
    return r >> 16;
}
static __device__ __forceinline__ unsigned int pack_bf16x2(float lo, float hi) {
    return f32_to_bf16(lo) | (f32_to_bf16(hi) << 16);
}
static __device__ __forceinline__ float bf16_lo(unsigned int w) {
    return __builtin_bit_cast(float, w << 16);
}
static __device__ __forceinline__ float bf16_hi(unsigned int w) {
    return __builtin_bit_cast(float, w & 0xFFFF0000u);
}
// low 32 bits of a generic LDS pointer == LDS byte address (ISA aperture rule)
static __device__ __forceinline__ unsigned lds_off(const void* p) {
    return (unsigned)(unsigned long long)(uintptr_t)p;
}
// CDNA5 async DMA: global -> LDS, 16B per lane, tracked by ASYNCcnt
static __device__ __forceinline__ void async_b128(unsigned lds, unsigned long long ga) {
    asm volatile("global_load_async_to_lds_b128 %0, %1, off"
                 :: "v"(lds), "v"(ga) : "memory");
}

// ---------------- graph prep ----------------
__global__ void count_edges(const int* __restrict__ ei, const int* __restrict__ et,
                            float* __restrict__ cnt, int E) {
    int e = blockIdx.x * blockDim.x + threadIdx.x;
    if (e >= E) return;
    int dst = ei[E + e];
    int r   = et[e];
    atomicAdd(cnt + dst * R_REL + r, 1.0f);
}

__global__ void invert_counts(float* __restrict__ cnt, int n) {
    int i = blockIdx.x * blockDim.x + threadIdx.x;
    if (i >= n) return;
    cnt[i] = 1.0f / fmaxf(cnt[i], 1.0f);
}

// f32 -> bf16 bulk convert (4 elems / thread)
__global__ void convert_bf16(const float* __restrict__ in, unsigned short* __restrict__ out, int n4) {
    int i = blockIdx.x * blockDim.x + threadIdx.x;
    if (i >= n4) return;
    float4 v = ((const float4*)in)[i];
    uint2 o;
    o.x = pack_bf16x2(v.x, v.y);
    o.y = pack_bf16x2(v.z, v.w);
    ((uint2*)out)[i] = o;
}

// one thread = 8 consecutive features of one edge (96 threads / edge), bf16 source
__global__ void scatter_feats(const int* __restrict__ ei, const int* __restrict__ et,
                              const unsigned short* __restrict__ feat,
                              float* __restrict__ sums, int E) {
    long long tid = (long long)blockIdx.x * blockDim.x + threadIdx.x;
    int e = (int)(tid / 96);
    int c = (int)(tid % 96);
    if (e >= E) return;
    int src = ei[e];
    int dst = ei[E + e];
    int r   = et[e];
    uint4 u = *(const uint4*)(feat + (size_t)src * D_IN + c * 8);
    float* s = sums + (size_t)(dst * R_REL + r) * D_IN + c * 8;
    atomicAdd(s + 0, bf16_lo(u.x)); atomicAdd(s + 1, bf16_hi(u.x));
    atomicAdd(s + 2, bf16_lo(u.y)); atomicAdd(s + 3, bf16_hi(u.y));
    atomicAdd(s + 4, bf16_lo(u.z)); atomicAdd(s + 5, bf16_hi(u.z));
    atomicAdd(s + 6, bf16_lo(u.w)); atomicAdd(s + 7, bf16_hi(u.w));
}

// mean + convert: abf[i] = bf16(sums[i] * cnt_inv[i/768])
__global__ void scale_convert(const float* __restrict__ sums, const float* __restrict__ cnt_inv,
                              unsigned short* __restrict__ abf, int n4) {
    int i = blockIdx.x * blockDim.x + threadIdx.x;
    if (i >= n4) return;
    float s = cnt_inv[i / 192];                 // (i*4)/768
    float4 v = ((const float4*)sums)[i];
    uint2 o;
    o.x = pack_bf16x2(v.x * s, v.y * s);
    o.y = pack_bf16x2(v.z * s, v.w * s);
    ((uint2*)abf)[i] = o;
}

// WT[o][k] = bf16( k < 6144 ? W[k/768][k%768][o] : root[k-6144][o] )
__global__ void prep_weights(const float* __restrict__ W, const float* __restrict__ root,
                             unsigned short* __restrict__ WT, int total) {
    int i = blockIdx.x * blockDim.x + threadIdx.x;
    if (i >= total) return;
    int o = i / K_AUG;
    int k = i - o * K_AUG;
    float v;
    if (k < K_SUM) {
        int r = k / D_IN, ii = k - r * D_IN;
        v = W[((size_t)r * D_IN + ii) * D_OUT + o];
    } else {
        v = root[(size_t)(k - K_SUM) * D_OUT + o];
    }
    WT[(size_t)o * K_AUG + k] = (unsigned short)f32_to_bf16(v);
}

// ---------------- WMMA GEMM ----------------
// C[n_nodes x 768] = [abf | dense] (bf16, [n x 6912]) * WT^T   (+bias, opt ReLU)
// WT stored transposed [768 x 6912] bf16 -> both operands K-contiguous.
// LDS staged with global_load_async_to_lds_b128, double-buffered.
__global__ __launch_bounds__(256)
void rgcn_gemm(const unsigned short* __restrict__ abf, const unsigned short* __restrict__ dense,
               const unsigned short* __restrict__ WT, const float* __restrict__ bias,
               float* __restrict__ outF, unsigned short* __restrict__ outB,
               int n_nodes, int relu) {
    __shared__ __align__(16) unsigned short As[2][BM * LDT];
    __shared__ __align__(16) unsigned short Bs[2][BN * LDT];

    const int tid  = threadIdx.x;
    const int wave = tid >> 5;
    const int lane = tid & 31;
    const int lh   = lane >> 4;      // lane half (0/1)
    const int lr   = lane & 15;
    const int wm   = wave >> 1;      // 0..3 -> M offset wm*32
    const int wn   = wave & 1;       // 0..1 -> N offset wn*64

    const int m0 = blockIdx.y * BM;
    const int n0 = blockIdx.x * BN;

    // staging map: thread owns two 16B chunks of one 64B row in each tile
    const int srow = tid >> 1;       // 0..127
    const int sq   = (tid & 1) * 2;  // chunk 0/2 (each chunk = 16B = 8 bf16)

    int gmS = m0 + srow; if (gmS >= n_nodes) gmS = n_nodes - 1;   // clamp (stores masked)
    const int gnS = n0 + srow;                                    // always < 768

    v8f acc[2][4];
#pragma unroll
    for (int mi = 0; mi < 2; ++mi)
#pragma unroll
        for (int ni = 0; ni < 4; ++ni)
#pragma unroll
            for (int q = 0; q < 8; ++q) acc[mi][ni][q] = 0.0f;

    auto stage = [&](int kt, int b) {
        const int k0 = kt * BK;
        const unsigned short* asrc = (k0 < K_SUM)
            ? abf   + (size_t)gmS * K_SUM + k0
            : dense + (size_t)gmS * D_IN  + (k0 - K_SUM);
        const unsigned short* bsrc = WT + (size_t)gnS * K_AUG + k0;
        unsigned aL = lds_off(&As[b][srow * LDT + sq * 8]);
        unsigned bL = lds_off(&Bs[b][srow * LDT + sq * 8]);
        async_b128(aL,      (unsigned long long)(uintptr_t)(asrc + sq * 8));
        async_b128(aL + 16, (unsigned long long)(uintptr_t)(asrc + sq * 8 + 8));
        async_b128(bL,      (unsigned long long)(uintptr_t)(bsrc + sq * 8));
        async_b128(bL + 16, (unsigned long long)(uintptr_t)(bsrc + sq * 8 + 8));
    };

    stage(0, 0);
    for (int kt = 0; kt < KT_TOT; ++kt) {
        const int cur = kt & 1;
        if (kt + 1 < KT_TOT) {
            stage(kt + 1, cur ^ 1);
            asm volatile("s_wait_asynccnt 4" ::: "memory");   // tile kt landed
        } else {
            asm volatile("s_wait_asynccnt 0" ::: "memory");
        }
        __syncthreads();

        // ---- fragments (ISA 16-bit A/B VGPR layouts) ----
        v16bf afrag[2], bfrag[4];
#pragma unroll
        for (int mi = 0; mi < 2; ++mi) {
            int row = wm * 32 + mi * 16 + lr;
            const U128* p = (const U128*)&As[cur][row * LDT];
            union { U128 u[2]; v16bf v; } q;
            q.u[0] = p[lh];          // K = 8*lh    .. 8*lh+7
            q.u[1] = p[2 + lh];      // K = 16+8*lh .. 16+8*lh+7
            afrag[mi] = q.v;
        }
#pragma unroll
        for (int ni = 0; ni < 4; ++ni) {
            int row = wn * 64 + ni * 16 + lr;
            const U128* p = (const U128*)&Bs[cur][row * LDT];
            union { U128 u[2]; v16bf v; } q;
            q.u[0] = p[lh];          // K = 8*lh    .. 8*lh+7
            q.u[1] = p[2 + lh];      // K = 16+8*lh .. 16+8*lh+7
            bfrag[ni] = q.v;
        }
#pragma unroll
        for (int mi = 0; mi < 2; ++mi)
#pragma unroll
            for (int ni = 0; ni < 4; ++ni)
                acc[mi][ni] = __builtin_amdgcn_wmma_f32_16x16x32_bf16(
                    false, afrag[mi], false, bfrag[ni],
                    (short)0, acc[mi][ni], false, false);

        if (kt + 1 < KT_TOT) __syncthreads();   // buf[cur] free for tile kt+2
    }

    // ---- epilogue: bias (+ReLU); C layout: VGPR v -> row v+8*lh, col lr ----
    const bool full = (m0 + BM) <= n_nodes;
#pragma unroll
    for (int mi = 0; mi < 2; ++mi)
#pragma unroll
        for (int ni = 0; ni < 4; ++ni) {
            const int col = n0 + wn * 64 + ni * 16 + lr;
            const float bcol = bias[col];
            const int rbase = m0 + wm * 32 + mi * 16 + 8 * lh;
            if (full) {
#pragma unroll
                for (int v = 0; v < 8; ++v) {
                    float x = acc[mi][ni][v] + bcol;
                    if (relu) x = fmaxf(x, 0.0f);
                    size_t idx = (size_t)(rbase + v) * D_OUT + col;
                    if (outB) outB[idx] = (unsigned short)f32_to_bf16(x);
                    else      outF[idx] = x;
                }
            } else {
#pragma unroll
                for (int v = 0; v < 8; ++v) {
                    if (rbase + v < n_nodes) {
                        float x = acc[mi][ni][v] + bcol;
                        if (relu) x = fmaxf(x, 0.0f);
                        size_t idx = (size_t)(rbase + v) * D_OUT + col;
                        if (outB) outB[idx] = (unsigned short)f32_to_bf16(x);
                        else      outF[idx] = x;
                    }
                }
            }
        }
}

// ---------------- host ----------------
extern "C" void kernel_launch(void* const* d_in, const int* in_sizes, int n_in,
                              void* d_out, int out_size, void* d_ws, size_t ws_size,
                              hipStream_t stream) {
    const float* x  = (const float*)d_in[0];
    const int*   ei = (const int*)d_in[1];
    const int*   et = (const int*)d_in[2];
    const float* W1 = (const float*)d_in[3];
    const float* r1 = (const float*)d_in[4];
    const float* b1 = (const float*)d_in[5];
    const float* W2 = (const float*)d_in[6];
    const float* r2 = (const float*)d_in[7];
    const float* b2 = (const float*)d_in[8];
    float* out = (float*)d_out;

    const int N = in_sizes[0] / D_IN;
    const int E = in_sizes[2];

    char* ws = (char*)d_ws;
    size_t off = 0;
    auto wsAlloc = [&](size_t bytes) -> char* {
        char* p = ws + off;
        off = (off + bytes + 255) & ~(size_t)255;
        return p;
    };
    float*          sums = (float*)wsAlloc((size_t)N * K_SUM * sizeof(float));   // 491.5 MB
    unsigned short* abf  = (unsigned short*)wsAlloc((size_t)N * K_SUM * 2);      // 245.8 MB
    float*          cnt  = (float*)wsAlloc((size_t)N * R_REL * sizeof(float));
    unsigned short* WT1  = (unsigned short*)wsAlloc((size_t)D_OUT * K_AUG * 2);
    unsigned short* WT2  = (unsigned short*)wsAlloc((size_t)D_OUT * K_AUG * 2);
    unsigned short* xbf  = (unsigned short*)wsAlloc((size_t)N * D_IN * 2);
    unsigned short* hbf  = (unsigned short*)wsAlloc((size_t)N * D_OUT * 2);

    const int wtot = D_OUT * K_AUG;
    const long long sth = (long long)E * 96;
    const int n4s = (int)((size_t)N * K_SUM / 4);
    const int n4x = (int)((size_t)N * D_IN / 4);
    dim3 gemmGrid(D_OUT / BN, (N + BM - 1) / BM);

    // graph/weight prep (shared by both layers)
    hipMemsetAsync(cnt, 0, (size_t)N * R_REL * sizeof(float), stream);
    count_edges<<<(E + 255) / 256, 256, 0, stream>>>(ei, et, cnt, E);
    invert_counts<<<(N * R_REL + 255) / 256, 256, 0, stream>>>(cnt, N * R_REL);
    prep_weights<<<(wtot + 255) / 256, 256, 0, stream>>>(W1, r1, WT1, wtot);
    prep_weights<<<(wtot + 255) / 256, 256, 0, stream>>>(W2, r2, WT2, wtot);
    convert_bf16<<<(n4x + 255) / 256, 256, 0, stream>>>(x, xbf, n4x);

    // layer 1: aggregate x -> means (f32 atomics) -> bf16 A -> WMMA GEMM -> hbf (ReLU)
    hipMemsetAsync(sums, 0, (size_t)N * K_SUM * sizeof(float), stream);
    scatter_feats<<<(int)((sth + 255) / 256), 256, 0, stream>>>(ei, et, xbf, sums, E);
    scale_convert<<<(n4s + 255) / 256, 256, 0, stream>>>(sums, cnt, abf, n4s);
    rgcn_gemm<<<gemmGrid, 256, 0, stream>>>(abf, xbf, WT1, b1, nullptr, hbf, N, 1);

    // layer 2: aggregate h -> means -> bf16 A -> WMMA GEMM -> out (f32)
    hipMemsetAsync(sums, 0, (size_t)N * K_SUM * sizeof(float), stream);
    scatter_feats<<<(int)((sth + 255) / 256), 256, 0, stream>>>(ei, et, hbf, sums, E);
    scale_convert<<<(n4s + 255) / 256, 256, 0, stream>>>(sums, cnt, abf, n4s);
    rgcn_gemm<<<gemmGrid, 256, 0, stream>>>(abf, hbf, WT2, b2, out, nullptr, N, 0);
}
